// TemporalEdgeGNN_85744727097866
// MI455X (gfx1250) — matrix operand order repaired and checked by reading, not compiled, not run on hardware
//
#include <hip/hip_runtime.h>
#include <hip/hip_bf16.h>

typedef _Float16 h16v __attribute__((ext_vector_type(16)));
typedef _Float16 h8v  __attribute__((ext_vector_type(8)));
typedef _Float16 h4v  __attribute__((ext_vector_type(4)));
typedef _Float16 h2v  __attribute__((ext_vector_type(2)));
typedef float    f32x8 __attribute__((ext_vector_type(8)));

#define DEVINL __device__ __forceinline__

constexpr int SEQ  = 20;
constexpr int IDIM = 32;
constexpr int HDIM = 128;

DEVINL f32x8 wmma_f16f32(h16v a, h16v b, f32x8 c) {
  return __builtin_amdgcn_wmma_f32_16x16x32_f16(false, a, false, b, (short)0, c,
                                                false, false);
}

// fast transcendentals: gfx1250 has v_tanh_f32; raw v_rcp_f32 for sigmoid
#if __has_builtin(__builtin_amdgcn_tanhf)
DEVINL float fast_tanh(float x) { return __builtin_amdgcn_tanhf(x); }
#else
DEVINL float fast_tanh(float x) {
  return 1.0f - 2.0f * __builtin_amdgcn_rcpf(__expf(2.0f * x) + 1.0f);
}
#endif
DEVINL float fast_sigmoid(float x) {
  return __builtin_amdgcn_rcpf(1.0f + __expf(-x));
}

// A fragment (16x32 f16, MxK). Per-lane data is two contiguous 8-element runs:
//   a[0..7]  = A[lane&15][kbase + 8*(lane>>4) + 0..7]
//   a[8..15] = A[lane&15][kbase + 16 + 8*(lane>>4) + 0..7]
// -> two 16B vector loads instead of 16 scalar u16 loads.
DEVINL h16v load_a_frag(const _Float16* src, int ld, int lane, int kbase) {
  const _Float16* p = src + (lane & 15) * ld + kbase + ((lane >> 4) << 3);
  h8v lo = *(const h8v*)p;
  h8v hi = *(const h8v*)(p + 16);
  h16v a;
#pragma unroll
  for (int j = 0; j < 8; ++j) { a[j] = lo[j]; a[j + 8] = hi[j]; }
  return a;
}

// B fragment: pre-swizzled, contiguous 16 f16 per lane per tile
DEVINL h16v load_b_frag(const _Float16* frags, int tileIdx, int lane) {
  return *(const h16v*)(frags + (((size_t)tileIdx * 32 + lane) << 4));
}

// ---------- weight packing: W[Nout][K] row-major -> B-fragment tiles ----------
__global__ void pack_b_kernel(const float* __restrict__ W, _Float16* __restrict__ out,
                              int K, int KT, int total) {
  int t = blockIdx.x * 256 + threadIdx.x;
  if (t >= total) return;
  int j    = t & 15;
  int lane = (t >> 4) & 31;
  int tile = t >> 9;
  int tk = tile % KT;
  int tn = tile / KT;
  int k = (tk << 5) + ((lane >> 4) << 3) + (j & 7) + ((j >> 3) << 4);
  int n = (tn << 4) + (lane & 15);
  out[t] = (_Float16)W[(size_t)n * K + k];
}

// ---------- degree / dinv ----------
__global__ void zero_f32_kernel(float* __restrict__ p, long n) {
  long t = (long)blockIdx.x * 256 + threadIdx.x;
  if (t < n) p[t] = 0.0f;
}
__global__ void zero_f32x4_kernel(float4* __restrict__ p, long n4) {
  long t = (long)blockIdx.x * 256 + threadIdx.x;
  if (t < n4) p[t] = make_float4(0.0f, 0.0f, 0.0f, 0.0f);
}
__global__ void deg_scatter_kernel(const int* __restrict__ dst, float* __restrict__ deg, long E) {
  long t = (long)blockIdx.x * 256 + threadIdx.x;
  if (t < E) atomicAdd(&deg[dst[t]], 1.0f);
}
__global__ void deg_finalize_kernel(float* __restrict__ deg, int N) {
  int t = blockIdx.x * 256 + threadIdx.x;
  if (t < N) deg[t] = rsqrtf(deg[t] + 1.0f);
}

// ---------- fused GRU over time: 16 nodes/block, 8 waves x 3 col-tiles ----------
__global__ __launch_bounds__(256) void gru_kernel(
    const float* __restrict__ x, const float* __restrict__ bih, const float* __restrict__ bhh,
    const _Float16* __restrict__ wihF, const _Float16* __restrict__ whhF,
    _Float16* __restrict__ hOut, int N) {
  __shared__ __align__(16) _Float16 xs[16 * IDIM];
  __shared__ __align__(16) _Float16 hs[16 * HDIM];
  __shared__ __align__(16) float grz[16 * 256];
  __shared__ __align__(16) float gxn[16 * HDIM];
  __shared__ __align__(16) float ghn[16 * HDIM];

  const int tid  = threadIdx.x;
  const int lane = tid & 31;
  const int w    = tid >> 5;
  const int base = blockIdx.x * 16;
  const int mrow = (lane >> 4) << 3;

  // elementwise ownership: row em, 8 consecutive cols starting at eq
  const int em = tid >> 4;
  const int eq = (tid & 15) << 3;

  float hreg[8];
#pragma unroll
  for (int i = 0; i < 8; ++i) hreg[i] = 0.0f;
  {
    h8v z;
#pragma unroll
    for (int i = 0; i < 8; ++i) z[i] = (_Float16)0.0f;
    *(h8v*)(hs + em * HDIM + eq) = z;
  }

  // hoist per-wave bias scalars out of the time loop
  float brz[3], bxn[3], bhn[3];
#pragma unroll
  for (int c3 = 0; c3 < 3; ++c3) {
    const int ct  = w * 3 + c3;
    const int col = (ct << 4) + (lane & 15);
    if (ct < 16) {
      brz[c3] = bih[col] + bhh[col];
      bxn[c3] = 0.0f; bhn[c3] = 0.0f;
    } else {
      brz[c3] = 0.0f;
      bxn[c3] = bih[col];
      bhn[c3] = bhh[col];
    }
  }
  __syncthreads();

  for (int t = 0; t < SEQ; ++t) {
    // stage x[:,t,:] tile as f16: thread -> (row tid>>4, 2 cols)
    {
      int m = tid >> 4, i2 = (tid & 15) << 1;
      int node = base + m;
      float2 v = make_float2(0.0f, 0.0f);
      if (node < N)
        v = *(const float2*)(x + ((size_t)node * SEQ + t) * IDIM + i2);
      h2v hx; hx[0] = (_Float16)v.x; hx[1] = (_Float16)v.y;
      *(h2v*)(xs + m * IDIM + i2) = hx;
    }
    __syncthreads();

    h16v ax  = load_a_frag(xs, IDIM, lane, 0);
    h16v ah0 = load_a_frag(hs, HDIM, lane, 0);
    h16v ah1 = load_a_frag(hs, HDIM, lane, 32);
    h16v ah2 = load_a_frag(hs, HDIM, lane, 64);
    h16v ah3 = load_a_frag(hs, HDIM, lane, 96);

#pragma unroll
    for (int c3 = 0; c3 < 3; ++c3) {
      const int ct  = w * 3 + c3;           // 0..23 gate column tile
      const int col = (ct << 4) + (lane & 15);
      if (ct < 16) {                        // r,z gates: x-part + h-part combined
        f32x8 c;
        float b0 = brz[c3];
#pragma unroll
        for (int r = 0; r < 8; ++r) c[r] = b0;
        c = wmma_f16f32(ax,  load_b_frag(wihF, ct,         lane), c);
        c = wmma_f16f32(ah0, load_b_frag(whhF, ct * 4 + 0, lane), c);
        c = wmma_f16f32(ah1, load_b_frag(whhF, ct * 4 + 1, lane), c);
        c = wmma_f16f32(ah2, load_b_frag(whhF, ct * 4 + 2, lane), c);
        c = wmma_f16f32(ah3, load_b_frag(whhF, ct * 4 + 3, lane), c);
#pragma unroll
        for (int r = 0; r < 8; ++r) grz[(mrow + r) * 256 + col] = c[r];
      } else {                              // n gate: keep x and h parts separate
        const int q = col - 256;
        f32x8 cx;
        float bx = bxn[c3];
#pragma unroll
        for (int r = 0; r < 8; ++r) cx[r] = bx;
        cx = wmma_f16f32(ax, load_b_frag(wihF, ct, lane), cx);
#pragma unroll
        for (int r = 0; r < 8; ++r) gxn[(mrow + r) * HDIM + q] = cx[r];
        f32x8 ch;
        float bh = bhn[c3];
#pragma unroll
        for (int r = 0; r < 8; ++r) ch[r] = bh;
        ch = wmma_f16f32(ah0, load_b_frag(whhF, ct * 4 + 0, lane), ch);
        ch = wmma_f16f32(ah1, load_b_frag(whhF, ct * 4 + 1, lane), ch);
        ch = wmma_f16f32(ah2, load_b_frag(whhF, ct * 4 + 2, lane), ch);
        ch = wmma_f16f32(ah3, load_b_frag(whhF, ct * 4 + 3, lane), ch);
#pragma unroll
        for (int r = 0; r < 8; ++r) ghn[(mrow + r) * HDIM + q] = ch[r];
      }
    }
    __syncthreads();

    // elementwise gate math + state update: contiguous 8-col slice per thread
    {
      const float* gr = grz + em * 256 + eq;
      const float* gx = gxn + em * HDIM + eq;
      const float* gh = ghn + em * HDIM + eq;
      h8v hv;
#pragma unroll
      for (int i = 0; i < 8; ++i) {
        float rr = fast_sigmoid(gr[i]);
        float zz = fast_sigmoid(gr[128 + i]);
        float nn = fast_tanh(gx[i] + rr * gh[i]);
        float hn = (1.0f - zz) * nn + zz * hreg[i];
        hreg[i] = hn;
        hv[i] = (_Float16)hn;
      }
      *(h8v*)(hs + em * HDIM + eq) = hv;
    }
    __syncthreads();
  }

  {
    int node = base + em;
    if (node < N) {
      h8v hv;
#pragma unroll
      for (int i = 0; i < 8; ++i) hv[i] = (_Float16)hreg[i];
      *(h8v*)(hOut + (size_t)node * HDIM + eq) = hv;
    }
  }
}

// ---------- GCN: y = (h @ W^T) * dinv[row] ----------
__global__ __launch_bounds__(256) void gcn_xw_kernel(
    const _Float16* __restrict__ hin, const _Float16* __restrict__ wF,
    const float* __restrict__ dinv, float* __restrict__ y, int N) {
  __shared__ __align__(16) _Float16 st[8][16 * HDIM];
  const int tid = threadIdx.x, lane = tid & 31, w = tid >> 5;
  const int tile = blockIdx.x * 8 + w;
  const int base = tile * 16;
  if (base >= N) return;
  _Float16* my = st[w];
  // stage 16 rows x 128 f16 via 16B vector loads (LDS same-wave in-order)
#pragma unroll
  for (int i = 0; i < 8; ++i) {
    int q = i * 32 + lane;           // 256 chunks of 16B
    int row = q >> 4, cof = q & 15;
    int node = base + row; if (node >= N) node = N - 1;
    *(float4*)(my + row * HDIM + cof * 8) =
        *(const float4*)(hin + (size_t)node * HDIM + cof * 8);
  }
  h16v a0 = load_a_frag(my, HDIM, lane, 0);
  h16v a1 = load_a_frag(my, HDIM, lane, 32);
  h16v a2 = load_a_frag(my, HDIM, lane, 64);
  h16v a3 = load_a_frag(my, HDIM, lane, 96);
  const int mrow = (lane >> 4) << 3;
  float dvr[8];
#pragma unroll
  for (int r = 0; r < 8; ++r) {
    int node = base + mrow + r; if (node >= N) node = N - 1;
    dvr[r] = dinv[node];
  }
#pragma unroll
  for (int ct = 0; ct < 8; ++ct) {
    f32x8 c;
#pragma unroll
    for (int r = 0; r < 8; ++r) c[r] = 0.0f;
    c = wmma_f16f32(a0, load_b_frag(wF, ct * 4 + 0, lane), c);
    c = wmma_f16f32(a1, load_b_frag(wF, ct * 4 + 1, lane), c);
    c = wmma_f16f32(a2, load_b_frag(wF, ct * 4 + 2, lane), c);
    c = wmma_f16f32(a3, load_b_frag(wF, ct * 4 + 3, lane), c);
    int col = (ct << 4) + (lane & 15);
#pragma unroll
    for (int r = 0; r < 8; ++r) {
      int node = base + mrow + r;
      if (node < N) y[(size_t)node * HDIM + col] = c[r] * dvr[r];
    }
  }
}

// ---------- edge scatter: s[dst] += y[src] ----------
__global__ void edge_scatter_kernel(const int* __restrict__ src, const int* __restrict__ dst,
                                    const float* __restrict__ y, float* __restrict__ s, long E) {
  long t = (long)blockIdx.x * 256 + threadIdx.x;
  if (t >= E * HDIM) return;
  long e = t >> 7;
  int col = (int)(t & 127);
  atomicAdd(&s[(size_t)dst[e] * HDIM + col], y[(size_t)src[e] * HDIM + col]);
}

// ---------- out = relu(dinv*(s+y) + b) -> f16 (4 elems/thread) ----------
__global__ void gcn_finish_kernel(const float* __restrict__ s, const float* __restrict__ y,
                                  const float* __restrict__ dinv, const float* __restrict__ b,
                                  _Float16* __restrict__ out, long n4) {
  long t = (long)blockIdx.x * 256 + threadIdx.x;
  if (t >= n4) return;
  long base = t * 4;
  long m = base >> 7;
  int col = (int)(base & 127);
  float4 sv = *(const float4*)(s + base);
  float4 yv = *(const float4*)(y + base);
  float4 bv = *(const float4*)(b + col);
  float d = dinv[m];
  h4v o;
  o[0] = (_Float16)fmaxf(d * (sv.x + yv.x) + bv.x, 0.0f);
  o[1] = (_Float16)fmaxf(d * (sv.y + yv.y) + bv.y, 0.0f);
  o[2] = (_Float16)fmaxf(d * (sv.z + yv.z) + bv.z, 0.0f);
  o[3] = (_Float16)fmaxf(d * (sv.w + yv.w) + bv.w, 0.0f);
  *(h4v*)(out + base) = o;
}

// ---------- fused edge MLP head: 16 edges per wave ----------
__global__ __launch_bounds__(128) void edge_mlp_kernel(
    const _Float16* __restrict__ h2, const int* __restrict__ src, const int* __restrict__ dst,
    const _Float16* __restrict__ w1F, const float* __restrict__ b1,
    const float* __restrict__ w2, const float* __restrict__ b2v,
    float* __restrict__ out, long E) {
  __shared__ __align__(16) _Float16 ef[4][16 * 256];
  const int tid = threadIdx.x, lane = tid & 31, w = tid >> 5;
  const long ebase = (long)blockIdx.x * 64 + (long)w * 16;
  _Float16* my = ef[w];
  // stage ef = [h2[src] | h2[dst]] : 16 rows x 512B, 16B chunks
#pragma unroll
  for (int i = 0; i < 16; ++i) {
    int q = i * 32 + lane;            // 512 chunks
    int row = q >> 5, cof = q & 31;
    long e = ebase + row; if (e >= E) e = E - 1;
    const _Float16* sp = (cof < 16)
        ? h2 + (size_t)src[e] * HDIM + cof * 8
        : h2 + (size_t)dst[e] * HDIM + (cof - 16) * 8;
    *(float4*)(my + row * 256 + cof * 8) = *(const float4*)sp;
  }
  h16v a[8];
#pragma unroll
  for (int kt = 0; kt < 8; ++kt) a[kt] = load_a_frag(my, 256, lane, kt * 32);

  const int mrow = (lane >> 4) << 3;
  float p[8];
#pragma unroll
  for (int r = 0; r < 8; ++r) p[r] = 0.0f;

#pragma unroll
  for (int ct = 0; ct < 8; ++ct) {
    int col = (ct << 4) + (lane & 15);
    f32x8 c;
    float bi = b1[col];
#pragma unroll
    for (int r = 0; r < 8; ++r) c[r] = bi;
#pragma unroll
    for (int kt = 0; kt < 8; ++kt)
      c = wmma_f16f32(a[kt], load_b_frag(w1F, ct * 8 + kt, lane), c);
    float w2c = w2[col];
#pragma unroll
    for (int r = 0; r < 8; ++r) p[r] += fmaxf(c[r], 0.0f) * w2c;
  }
  // reduce over the 16 lanes holding the same rows (xor masks stay within half-wave)
#pragma unroll
  for (int r = 0; r < 8; ++r) {
    float v = p[r];
    v += __shfl_xor(v, 1);
    v += __shfl_xor(v, 2);
    v += __shfl_xor(v, 4);
    v += __shfl_xor(v, 8);
    p[r] = v;
  }
  if ((lane & 15) == 0) {
    float bb = b2v[0];
#pragma unroll
    for (int r = 0; r < 8; ++r) {
      long e = ebase + mrow + r;
      if (e < E) out[e] = p[r] + bb;
    }
  }
}

extern "C" void kernel_launch(void* const* d_in, const int* in_sizes, int n_in,
                              void* d_out, int out_size, void* d_ws, size_t ws_size,
                              hipStream_t stream) {
  const float* x    = (const float*)d_in[0];
  const int*   ei   = (const int*)d_in[1];
  const float* Wih  = (const float*)d_in[2];
  const float* Whh  = (const float*)d_in[3];
  const float* bih  = (const float*)d_in[4];
  const float* bhh  = (const float*)d_in[5];
  const float* g1W  = (const float*)d_in[6];
  const float* g1b  = (const float*)d_in[7];
  const float* g2W  = (const float*)d_in[8];
  const float* g2b  = (const float*)d_in[9];
  const float* m1W  = (const float*)d_in[10];
  const float* m1b  = (const float*)d_in[11];
  const float* m2W  = (const float*)d_in[12];
  const float* m2b  = (const float*)d_in[13];
  float* outp = (float*)d_out;

  const int  N = in_sizes[0] / (SEQ * IDIM);
  const long E = in_sizes[1] / 2;
  const int* srcp = ei;
  const int* dstp = ei + E;

  char* ws = (char*)d_ws;
  size_t off = 0;
  auto alloc = [&](size_t bytes) -> void* {
    void* p = ws + off;
    off = (off + bytes + 255) & ~(size_t)255;
    return p;
  };
  _Float16* wihF = (_Float16*)alloc(24 * 512 * 2);
  _Float16* whhF = (_Float16*)alloc(96 * 512 * 2);
  _Float16* g1F  = (_Float16*)alloc(32 * 512 * 2);
  _Float16* g2F  = (_Float16*)alloc(32 * 512 * 2);
  _Float16* m1F  = (_Float16*)alloc(64 * 512 * 2);
  float*    dinv = (float*)alloc((size_t)N * 4);
  _Float16* hA   = (_Float16*)alloc((size_t)N * HDIM * 2);
  _Float16* hB   = (_Float16*)alloc((size_t)N * HDIM * 2);
  float*    ybuf = (float*)alloc((size_t)N * HDIM * 4);
  float*    sbuf = (float*)alloc((size_t)N * HDIM * 4);
  (void)ws_size; (void)n_in; (void)out_size;

  // pack weights into B-fragment layout (f16)
  pack_b_kernel<<<(12288 + 255) / 256, 256, 0, stream>>>(Wih, wihF, 32,  1, 12288);
  pack_b_kernel<<<(49152 + 255) / 256, 256, 0, stream>>>(Whh, whhF, 128, 4, 49152);
  pack_b_kernel<<<(16384 + 255) / 256, 256, 0, stream>>>(g1W, g1F,  128, 4, 16384);
  pack_b_kernel<<<(16384 + 255) / 256, 256, 0, stream>>>(g2W, g2F,  128, 4, 16384);
  pack_b_kernel<<<(32768 + 255) / 256, 256, 0, stream>>>(m1W, m1F,  256, 8, 32768);

  // symmetric-norm degree factors
  zero_f32_kernel<<<(N + 255) / 256, 256, 0, stream>>>(dinv, N);
  deg_scatter_kernel<<<(int)((E + 255) / 256), 256, 0, stream>>>(dstp, dinv, E);
  deg_finalize_kernel<<<(N + 255) / 256, 256, 0, stream>>>(dinv, N);

  // GRU -> hA (f16)
  gru_kernel<<<(N + 15) / 16, 256, 0, stream>>>(x, bih, bhh, wihF, whhF, hA, N);

  const long NH = (long)N * HDIM;
  const long NH4 = NH / 4;
  const int  xwBlocks = ((N + 15) / 16 + 7) / 8;
  const int  ewBlocks = (int)((E * HDIM + 255) / 256);
  const int  n4Blocks = (int)((NH4 + 255) / 256);

  // GCN layer 1: hA -> hB
  gcn_xw_kernel<<<xwBlocks, 256, 0, stream>>>(hA, g1F, dinv, ybuf, N);
  zero_f32x4_kernel<<<n4Blocks, 256, 0, stream>>>((float4*)sbuf, NH4);
  edge_scatter_kernel<<<ewBlocks, 256, 0, stream>>>(srcp, dstp, ybuf, sbuf, E);
  gcn_finish_kernel<<<n4Blocks, 256, 0, stream>>>(sbuf, ybuf, dinv, g1b, hB, NH4);

  // GCN layer 2: hB -> hA
  gcn_xw_kernel<<<xwBlocks, 256, 0, stream>>>(hB, g2F, dinv, ybuf, N);
  zero_f32x4_kernel<<<n4Blocks, 256, 0, stream>>>((float4*)sbuf, NH4);
  edge_scatter_kernel<<<ewBlocks, 256, 0, stream>>>(srcp, dstp, ybuf, sbuf, E);
  gcn_finish_kernel<<<n4Blocks, 256, 0, stream>>>(sbuf, ybuf, dinv, g2b, hA, NH4);

  // edge MLP head -> logits
  edge_mlp_kernel<<<(int)((E + 63) / 64), 128, 0, stream>>>(
      hA, srcp, dstp, m1F, m1b, m2W, m2b, outp, E);
}
